// PointTransformerLayer_32298154066756
// MI455X (gfx1250) — compile-verified, matrix operand order checked by reading, not run on hardware
//
// Point-Transformer layer for MI455X (gfx1250, wave32, WMMA bf16 16x16x32).
//
// Dataflow (memory-bound problem, 23.3 TB/s HBM, 192MB L2):
//   1) prep: swizzle w_q/w_k/w_v/w_w1/w_w2 into WMMA B-fragment order (bf16), zero stat accumulators
//   2) qkv_gemm: x@W for q,k,v via v_wmma_f32_16x16x32_bf16 (L2-resident outputs, 30MB each)
//   3) pos_stats + finalize: BatchNorm stats for position MLP (3 ch)
//   4) w1_stats + finalize: BN stats of (gk - xq + p_r) over N*NS (128 ch), p_r recomputed
//   5) w2_stats + finalize: BN stats of first attention MLP output via per-point WMMA [16x128]@[128x16]
//   6) attn_final: fused per-point pipeline: gather -> bn1 -> WMMA1 -> bn2 -> WMMA2 ->
//                  softmax over neighbors (shfl_xor 16) -> share-grouped weighted sum.
#include <hip/hip_runtime.h>
#include <hip/hip_bf16.h>

#define NPTS 60000
#define NSAMP 16
#define CH 128
#define NC (NPTS*CH)

typedef __attribute__((ext_vector_type(16))) __bf16 v16bf;
typedef __attribute__((ext_vector_type(8)))  __bf16 v8bf;
typedef __attribute__((ext_vector_type(2)))  __bf16 bf16x2;
typedef __attribute__((ext_vector_type(8)))  float  v8f;
typedef __attribute__((ext_vector_type(4)))  float  f32x4;
typedef __attribute__((ext_vector_type(4)))  unsigned u32x4;
typedef __attribute__((ext_vector_type(2)))  unsigned u32x2;

static __device__ __forceinline__ unsigned pkbf(float a, float b) {
  bf16x2 t; t[0] = (__bf16)a; t[1] = (__bf16)b;
  return __builtin_bit_cast(unsigned, t);
}

// K-offset (within a 32-wide K chunk) for fragment slot s (0..15) of a lane.
// Matches ISA 16-bit A/B layout: VGPR v holds K pair; lanes 16-31 shifted by 8;
// VGPRs 4-7 cover K+16.
static __device__ __forceinline__ int frag_koff(int lane, int s) {
  int v = s >> 1;
  return ((v >> 2) << 4) + ((lane >> 4) << 3) + ((v & 3) << 1) + (s & 1);
}

static __device__ __forceinline__ v16bf joinbf(u32x4 q0, u32x4 q1) {
  v8bf lo = __builtin_bit_cast(v8bf, q0);
  v8bf hi = __builtin_bit_cast(v8bf, q1);
  v16bf a;
#pragma unroll
  for (int i = 0; i < 8; i++) { a[i] = lo[i]; a[8 + i] = hi[i]; }
  return a;
}

// A fragment from LDS tile stored as [16 rows][64 bf16-pairs] (row-major K pairs).
static __device__ __forceinline__ v16bf load_afrag(const unsigned* base, int lane, int kc) {
  int row = lane & 15, hl = lane >> 4;
  const unsigned* pa = base + row * 64 + kc * 16 + hl * 4;
  u32x4 q0 = *(const u32x4*)pa;       // slots 0..7  (K group 0)
  u32x4 q1 = *(const u32x4*)(pa + 8); // slots 8..15 (K group 1, +16)
  return joinbf(q0, q1);
}

// B fragment pre-swizzled in global memory: 16 contiguous bf16 per lane.
static __device__ __forceinline__ v16bf load_bfrag(const __bf16* b) {
  u32x4 q0 = *(const u32x4*)b;
  u32x4 q1 = *(const u32x4*)(b + 8);
  return joinbf(q0, q1);
}

static __device__ __forceinline__ v8f wmma_bf16(v16bf a, v16bf b, v8f c) {
  return __builtin_amdgcn_wmma_f32_16x16x32_bf16(false, a, false, b, (short)0, c, false, false);
}

// ---------------------------------------------------------------------------
// Kernel 1: weight swizzle into fragment order + zero stats accumulators.
// ---------------------------------------------------------------------------
__global__ __launch_bounds__(256) void prep_kernel(
    const float* __restrict__ wq, const float* __restrict__ wk, const float* __restrict__ wv,
    const float* __restrict__ ww1, const float* __restrict__ ww2,
    __bf16* __restrict__ wqf, __bf16* __restrict__ wkf, __bf16* __restrict__ wvf,
    __bf16* __restrict__ w1f, __bf16* __restrict__ w2f,
    float* __restrict__ stats) {
  const int QKV = 3 * 16384;
  int t = blockIdx.x * 256 + threadIdx.x;
  if (t < QKV) {
    int s = t & 15, lane = (t >> 4) & 31, kc = (t >> 9) & 3, jt = (t >> 11) & 7, m = t >> 14;
    int k = kc * 32 + frag_koff(lane, s);
    int n = jt * 16 + (lane & 15);
    const float* w = (m == 0) ? wq : (m == 1) ? wk : wv;
    __bf16* d = (m == 0) ? wqf : (m == 1) ? wkf : wvf;
    d[t & 16383] = (__bf16)w[k * CH + n];
  } else if (t < QKV + 2048) {
    int u = t - QKV;
    int s = u & 15, lane = (u >> 4) & 31, kc = (u >> 9) & 3;
    int k = kc * 32 + frag_koff(lane, s);
    w1f[u] = (__bf16)ww1[k * 16 + (lane & 15)];
  } else if (t < QKV + 2048 + 512) {
    int u = t - QKV - 2048;
    int s = u & 15, lane = (u >> 4) & 31;
    int k = frag_koff(lane, s); // single K chunk, upper 16 zero-padded
    w2f[u] = (k < 16) ? (__bf16)ww2[k * 16 + (lane & 15)] : (__bf16)0.f;
  } else if (t < QKV + 2048 + 512 + 294) {
    stats[t - (QKV + 2048 + 512)] = 0.f;
  }
}

// ---------------------------------------------------------------------------
// Kernel 2: QKV GEMM. One block = 16 rows of x; 8 waves = 8 column tiles;
// each wave does q,k,v tiles (12 WMMAs).
// ---------------------------------------------------------------------------
__global__ __launch_bounds__(256) void qkv_gemm(
    const float* __restrict__ x,
    const float* __restrict__ bq, const float* __restrict__ bk, const float* __restrict__ bv,
    const __bf16* __restrict__ wqf, const __bf16* __restrict__ wkf, const __bf16* __restrict__ wvf,
    float* __restrict__ xq, float* __restrict__ xk, float* __restrict__ xv) {
  __shared__ unsigned lds[16 * 64]; // 16 rows x 128 bf16 (packed pairs)
  int row0 = blockIdx.x * 16;
  int t = threadIdx.x;
  {
    int pr0 = t * 4;
    int r = pr0 >> 6, kp = pr0 & 63;
    const float* src = &x[(row0 + r) * CH + kp * 2];
    f32x4 u0 = *(const f32x4*)src;
    f32x4 u1 = *(const f32x4*)(src + 4);
    u32x4 w;
    w[0] = pkbf(u0[0], u0[1]); w[1] = pkbf(u0[2], u0[3]);
    w[2] = pkbf(u1[0], u1[1]); w[3] = pkbf(u1[2], u1[3]);
    *(u32x4*)&lds[pr0] = w;
  }
  __syncthreads();
  int wave = t >> 5, lane = t & 31;
  int jt = wave;
  v8f aq = {}, ak = {}, av = {};
#pragma unroll
  for (int kc = 0; kc < 4; kc++) {
    v16bf A = load_afrag(lds, lane, kc);
    int boff = ((jt * 4 + kc) * 32 + lane) * 16;
    aq = wmma_bf16(A, load_bfrag(wqf + boff), aq);
    ak = wmma_bf16(A, load_bfrag(wkf + boff), ak);
    av = wmma_bf16(A, load_bfrag(wvf + boff), av);
  }
  int col = jt * 16 + (lane & 15), hl = lane >> 4;
  float cq = bq[col], ck = bk[col], cv = bv[col];
#pragma unroll
  for (int r = 0; r < 8; r++) {
    int m = row0 + r + 8 * hl;
    xq[m * CH + col] = aq[r] + cq;
    xk[m * CH + col] = ak[r] + ck;
    xv[m * CH + col] = av[r] + cv;
  }
}

// ---------------------------------------------------------------------------
// Kernel 3: stats (sum, sumsq) of gp@w_p1+b_p1 over N*NS, 3 channels.
// ---------------------------------------------------------------------------
__global__ __launch_bounds__(256) void pos_stats(
    const float* __restrict__ p, const int* __restrict__ idx,
    const float* __restrict__ wp1, const float* __restrict__ bp1,
    float* __restrict__ stats) {
  float a0 = wp1[0], a1 = wp1[1], a2 = wp1[2], a3 = wp1[3], a4 = wp1[4],
        a5 = wp1[5], a6 = wp1[6], a7 = wp1[7], a8 = wp1[8];
  float b0 = bp1[0], b1 = bp1[1], b2 = bp1[2];
  float s0 = 0, s1 = 0, s2 = 0, q0 = 0, q1 = 0, q2 = 0;
  int total = NPTS * NSAMP;
  for (int e = blockIdx.x * 256 + threadIdx.x; e < total; e += gridDim.x * 256) {
    int j = idx[e];
    const float* pj = p + 3 * j;
    float px = pj[0], py = pj[1], pz = pj[2];
    float y0 = fmaf(px, a0, fmaf(py, a3, fmaf(pz, a6, b0)));
    float y1 = fmaf(px, a1, fmaf(py, a4, fmaf(pz, a7, b1)));
    float y2 = fmaf(px, a2, fmaf(py, a5, fmaf(pz, a8, b2)));
    s0 += y0; s1 += y1; s2 += y2;
    q0 = fmaf(y0, y0, q0); q1 = fmaf(y1, y1, q1); q2 = fmaf(y2, y2, q2);
  }
  __shared__ float red[6];
  if (threadIdx.x < 6) red[threadIdx.x] = 0.f;
  __syncthreads();
  atomicAdd(&red[0], s0); atomicAdd(&red[1], s1); atomicAdd(&red[2], s2);
  atomicAdd(&red[3], q0); atomicAdd(&red[4], q1); atomicAdd(&red[5], q2);
  __syncthreads();
  if (threadIdx.x < 6) atomicAdd(&stats[threadIdx.x], red[threadIdx.x]);
}

// ---------------------------------------------------------------------------
// BN finalize: scale = g/sqrt(var+eps), shift = b - mean*scale.
// ---------------------------------------------------------------------------
__global__ void finalize_bn(const float* __restrict__ sum, const float* __restrict__ sq,
                            const float* __restrict__ g, const float* __restrict__ b,
                            float* __restrict__ scale, float* __restrict__ shift,
                            int nch, float invc) {
  int c = threadIdx.x;
  if (c >= nch) return;
  float m = sum[c] * invc;
  float v = sq[c] * invc - m * m;
  float sc = g[c] * rsqrtf(v + 1e-5f);
  scale[c] = sc;
  shift[c] = b[c] - m * sc;
}

// ---------------------------------------------------------------------------
// Kernel 4: stats of w_pre = gk - xq + p_r over N*NS, 128 channels.
// One wave per point iteration; lane owns channels 4*lane..4*lane+3.
// ---------------------------------------------------------------------------
__global__ __launch_bounds__(256) void w1_stats(
    const float* __restrict__ p, const int* __restrict__ idx,
    const float* __restrict__ xq, const float* __restrict__ xk,
    const float* __restrict__ wp1, const float* __restrict__ bp1,
    const float* __restrict__ wp2, const float* __restrict__ bp2,
    const float* __restrict__ pS, const float* __restrict__ pH,
    float* __restrict__ statSum, float* __restrict__ statSq) {
  int t = threadIdx.x, wave = t >> 5, lane = t & 31;
  int c0 = lane * 4;
  float a0 = wp1[0], a1 = wp1[1], a2 = wp1[2], a3 = wp1[3], a4 = wp1[4],
        a5 = wp1[5], a6 = wp1[6], a7 = wp1[7], a8 = wp1[8];
  float b0 = bp1[0], b1 = bp1[1], b2 = bp1[2];
  float ps0 = pS[0], ps1 = pS[1], ps2 = pS[2];
  float ph0 = pH[0], ph1 = pH[1], ph2 = pH[2];
  f32x4 w2c0 = *(const f32x4*)&wp2[c0];
  f32x4 w2c1 = *(const f32x4*)&wp2[CH + c0];
  f32x4 w2c2 = *(const f32x4*)&wp2[2 * CH + c0];
  f32x4 pb4  = *(const f32x4*)&bp2[c0];
  f32x4 sm = {0.f, 0.f, 0.f, 0.f}, sq = {0.f, 0.f, 0.f, 0.f};
  int W = gridDim.x * 8;
  for (int n = blockIdx.x * 8 + wave; n < NPTS; n += W) {
    f32x4 xq4 = *(const f32x4*)&xq[n * CH + c0];
    for (int s = 0; s < NSAMP; s++) {
      int j = idx[n * NSAMP + s];
      const float* pj = p + 3 * j;
      float px = pj[0], py = pj[1], pz = pj[2];
      float y0 = fmaxf(fmaf(fmaf(px, a0, fmaf(py, a3, fmaf(pz, a6, b0))), ps0, ph0), 0.f);
      float y1 = fmaxf(fmaf(fmaf(px, a1, fmaf(py, a4, fmaf(pz, a7, b1))), ps1, ph1), 0.f);
      float y2 = fmaxf(fmaf(fmaf(px, a2, fmaf(py, a5, fmaf(pz, a8, b2))), ps2, ph2), 0.f);
      f32x4 kk = *(const f32x4*)&xk[j * CH + c0];
#pragma unroll
      for (int i = 0; i < 4; i++) {
        float pr = fmaf(y0, w2c0[i], fmaf(y1, w2c1[i], fmaf(y2, w2c2[i], pb4[i])));
        float w = kk[i] - xq4[i] + pr;
        sm[i] += w;
        sq[i] = fmaf(w, w, sq[i]);
      }
    }
  }
  __shared__ float rs[128], rq[128];
  if (t < 128) { rs[t] = 0.f; rq[t] = 0.f; }
  __syncthreads();
#pragma unroll
  for (int i = 0; i < 4; i++) { atomicAdd(&rs[c0 + i], sm[i]); atomicAdd(&rq[c0 + i], sq[i]); }
  __syncthreads();
  if (t < 128) { atomicAdd(&statSum[t], rs[t]); atomicAdd(&statSq[t], rq[t]); }
}

// ---------------------------------------------------------------------------
// Kernel 5: stats of y1 = relu(bn1(w_pre)) @ w_w1 + b_w1 over N*NS, 16 channels.
// Per point: stage bn1(w_pre) as bf16 in LDS, one [16x128]@[128x16] via 4 WMMAs.
// ---------------------------------------------------------------------------
__global__ __launch_bounds__(256) void w2_stats(
    const float* __restrict__ p, const int* __restrict__ idx,
    const float* __restrict__ xq, const float* __restrict__ xk,
    const float* __restrict__ wp1, const float* __restrict__ bp1,
    const float* __restrict__ wp2, const float* __restrict__ bp2,
    const float* __restrict__ pS, const float* __restrict__ pH,
    const float* __restrict__ w1S, const float* __restrict__ w1H,
    const float* __restrict__ bw1,
    const __bf16* __restrict__ w1f,
    float* __restrict__ statSum, float* __restrict__ statSq) {
  __shared__ unsigned ldsA[8][1024];
  int t = threadIdx.x, wave = t >> 5, lane = t & 31;
  int c0 = lane * 4, cs = lane & 15;
  unsigned* la = &ldsA[wave][0];
  float a0 = wp1[0], a1 = wp1[1], a2 = wp1[2], a3 = wp1[3], a4 = wp1[4],
        a5 = wp1[5], a6 = wp1[6], a7 = wp1[7], a8 = wp1[8];
  float b0 = bp1[0], b1 = bp1[1], b2 = bp1[2];
  float ps0 = pS[0], ps1 = pS[1], ps2 = pS[2];
  float ph0 = pH[0], ph1 = pH[1], ph2 = pH[2];
  f32x4 w2c0 = *(const f32x4*)&wp2[c0];
  f32x4 w2c1 = *(const f32x4*)&wp2[CH + c0];
  f32x4 w2c2 = *(const f32x4*)&wp2[2 * CH + c0];
  f32x4 pb4  = *(const f32x4*)&bp2[c0];
  f32x4 s1v  = *(const f32x4*)&w1S[c0];
  f32x4 h1v  = *(const f32x4*)&w1H[c0];
  float bw = bw1[cs];
  float ssum = 0.f, ssq = 0.f;
  int W = gridDim.x * 8;
  for (int n = blockIdx.x * 8 + wave; n < NPTS; n += W) {
    f32x4 xq4 = *(const f32x4*)&xq[n * CH + c0];
    for (int s = 0; s < NSAMP; s++) {
      int j = idx[n * NSAMP + s];
      const float* pj = p + 3 * j;
      float px = pj[0], py = pj[1], pz = pj[2];
      float y0 = fmaxf(fmaf(fmaf(px, a0, fmaf(py, a3, fmaf(pz, a6, b0))), ps0, ph0), 0.f);
      float y1 = fmaxf(fmaf(fmaf(px, a1, fmaf(py, a4, fmaf(pz, a7, b1))), ps1, ph1), 0.f);
      float y2 = fmaxf(fmaf(fmaf(px, a2, fmaf(py, a5, fmaf(pz, a8, b2))), ps2, ph2), 0.f);
      f32x4 kk = *(const f32x4*)&xk[j * CH + c0];
      float z[4];
#pragma unroll
      for (int i = 0; i < 4; i++) {
        float pr = fmaf(y0, w2c0[i], fmaf(y1, w2c1[i], fmaf(y2, w2c2[i], pb4[i])));
        float w = kk[i] - xq4[i] + pr;
        z[i] = fmaxf(fmaf(w, s1v[i], h1v[i]), 0.f);
      }
      u32x2 pk; pk[0] = pkbf(z[0], z[1]); pk[1] = pkbf(z[2], z[3]);
      *(u32x2*)&la[s * 64 + lane * 2] = pk;
    }
    v8f acc = {};
#pragma unroll
    for (int kc = 0; kc < 4; kc++) {
      v16bf A = load_afrag(la, lane, kc);
      v16bf B = load_bfrag(w1f + (kc * 32 + lane) * 16);
      acc = wmma_bf16(A, B, acc);
    }
#pragma unroll
    for (int r = 0; r < 8; r++) {
      float y = acc[r] + bw;
      ssum += y;
      ssq = fmaf(y, y, ssq);
    }
  }
  __shared__ float rs[16], rq[16];
  if (t < 16) { rs[t] = 0.f; rq[t] = 0.f; }
  __syncthreads();
  atomicAdd(&rs[cs], ssum); atomicAdd(&rq[cs], ssq);
  __syncthreads();
  if (t < 16) { atomicAdd(&statSum[t], rs[t]); atomicAdd(&statSq[t], rq[t]); }
}

// ---------------------------------------------------------------------------
// Kernel 6: final fused attention. Per point: bn1 -> WMMA1 -> bn2 -> WMMA2 ->
// softmax over neighbors -> weighted sum with share-grouped broadcast.
// ---------------------------------------------------------------------------
__global__ __launch_bounds__(256) void attn_final(
    const float* __restrict__ p, const int* __restrict__ idx,
    const float* __restrict__ xq, const float* __restrict__ xk, const float* __restrict__ xv,
    const float* __restrict__ wp1, const float* __restrict__ bp1,
    const float* __restrict__ wp2, const float* __restrict__ bp2,
    const float* __restrict__ pS, const float* __restrict__ pH,
    const float* __restrict__ w1S, const float* __restrict__ w1H, const float* __restrict__ bw1,
    const float* __restrict__ w2S, const float* __restrict__ w2H, const float* __restrict__ bw2,
    const __bf16* __restrict__ w1f, const __bf16* __restrict__ w2f,
    float* __restrict__ out) {
  __shared__ unsigned ldsA[8][1024]; // bn1'd w_pre, bf16 pairs [16 s][64]
  __shared__ __bf16   ldsB[8][512];  // bn2'd y1 [16 s][32 k] (K padded with 0)
  __shared__ float    ldsW[8][256];  // softmax weights [16 s][16 cs]
  __shared__ float    ldsR[8][64];   // relu'd pos-MLP hidden [16 s][3(+1)]
  int t = threadIdx.x, wave = t >> 5, lane = t & 31;
  int c0 = lane * 4, cs = lane & 15, hl = lane >> 4;
  unsigned* la = &ldsA[wave][0];
  __bf16*   lb = &ldsB[wave][0];
  float*    lw = &ldsW[wave][0];
  float*    lr = &ldsR[wave][0];
  float a0 = wp1[0], a1 = wp1[1], a2 = wp1[2], a3 = wp1[3], a4 = wp1[4],
        a5 = wp1[5], a6 = wp1[6], a7 = wp1[7], a8 = wp1[8];
  float b0 = bp1[0], b1 = bp1[1], b2 = bp1[2];
  float ps0 = pS[0], ps1 = pS[1], ps2 = pS[2];
  float ph0 = pH[0], ph1 = pH[1], ph2 = pH[2];
  f32x4 w2c0 = *(const f32x4*)&wp2[c0];
  f32x4 w2c1 = *(const f32x4*)&wp2[CH + c0];
  f32x4 w2c2 = *(const f32x4*)&wp2[2 * CH + c0];
  f32x4 pb4  = *(const f32x4*)&bp2[c0];
  f32x4 s1v  = *(const f32x4*)&w1S[c0];
  f32x4 h1v  = *(const f32x4*)&w1H[c0];
  float sc2 = w2S[cs], hf2 = w2H[cs], bb1 = bw1[cs], bb2 = bw2[cs];
  // zero K=16..31 pad region of ldsB once (never overwritten)
#pragma unroll
  for (int i = 0; i < 8; i++) {
    int e = lane * 8 + i;
    lb[(e >> 4) * 32 + 16 + (e & 15)] = (__bf16)0.f;
  }
  int W = gridDim.x * 8;
  for (int n = blockIdx.x * 8 + wave; n < NPTS; n += W) {
    f32x4 xq4 = *(const f32x4*)&xq[n * CH + c0];
    for (int s = 0; s < NSAMP; s++) {
      int j = idx[n * NSAMP + s];
      const float* pj = p + 3 * j;
      float px = pj[0], py = pj[1], pz = pj[2];
      float y0 = fmaxf(fmaf(fmaf(px, a0, fmaf(py, a3, fmaf(pz, a6, b0))), ps0, ph0), 0.f);
      float y1 = fmaxf(fmaf(fmaf(px, a1, fmaf(py, a4, fmaf(pz, a7, b1))), ps1, ph1), 0.f);
      float y2 = fmaxf(fmaf(fmaf(px, a2, fmaf(py, a5, fmaf(pz, a8, b2))), ps2, ph2), 0.f);
      if (lane < 3) lr[s * 4 + lane] = (lane == 0) ? y0 : ((lane == 1) ? y1 : y2);
      f32x4 kk = *(const f32x4*)&xk[j * CH + c0];
      float z[4];
#pragma unroll
      for (int i = 0; i < 4; i++) {
        float pr = fmaf(y0, w2c0[i], fmaf(y1, w2c1[i], fmaf(y2, w2c2[i], pb4[i])));
        float w = kk[i] - xq4[i] + pr;
        z[i] = fmaxf(fmaf(w, s1v[i], h1v[i]), 0.f);
      }
      u32x2 pk; pk[0] = pkbf(z[0], z[1]); pk[1] = pkbf(z[2], z[3]);
      *(u32x2*)&la[s * 64 + lane * 2] = pk;
    }
    // WMMA1: [16 neighbors x 128] @ [128 x 16]
    v8f acc = {};
#pragma unroll
    for (int kc = 0; kc < 4; kc++) {
      v16bf A = load_afrag(la, lane, kc);
      v16bf B = load_bfrag(w1f + (kc * 32 + lane) * 16);
      acc = wmma_bf16(A, B, acc);
    }
    // bn2 + relu, stage as bf16 for WMMA2 (lane holds column cs, rows r+8*hl)
#pragma unroll
    for (int r = 0; r < 8; r++) {
      float y = acc[r] + bb1;
      lb[(r + 8 * hl) * 32 + cs] = (__bf16)fmaxf(fmaf(y, sc2, hf2), 0.f);
    }
    // WMMA2: [16 x 16(pad 32)] @ [16(pad 32) x 16]
    v16bf A2;
    {
      int row = lane & 15;
#pragma unroll
      for (int vv = 0; vv < 8; vv++) {
        int k = ((vv >> 2) << 4) + (hl << 3) + ((vv & 3) << 1);
        A2[2 * vv]     = lb[row * 32 + k];
        A2[2 * vv + 1] = lb[row * 32 + k + 1];
      }
    }
    v8f acc2 = {};
    acc2 = wmma_bf16(A2, load_bfrag(w2f + lane * 16), acc2);
    // softmax over the 16 neighbors of each column (split lane <-> lane^16)
    float yv[8];
    float mx = -3.0e38f;
#pragma unroll
    for (int r = 0; r < 8; r++) { yv[r] = acc2[r] + bb2; mx = fmaxf(mx, yv[r]); }
    mx = fmaxf(mx, __shfl_xor(mx, 16, 32));
    float se = 0.f;
#pragma unroll
    for (int r = 0; r < 8; r++) { yv[r] = __expf(yv[r] - mx); se += yv[r]; }
    se += __shfl_xor(se, 16, 32);
    float inv = 1.0f / se;
#pragma unroll
    for (int r = 0; r < 8; r++) lw[(r + 8 * hl) * 16 + cs] = yv[r] * inv;
    // weighted sum: out[c] = sum_s (gv + p_r) * w[s, c%16]; lane owns c0..c0+3
    f32x4 o = {0.f, 0.f, 0.f, 0.f};
    for (int s = 0; s < NSAMP; s++) {
      int j = idx[n * NSAMP + s];
      f32x4 gv = *(const f32x4*)&xv[j * CH + c0];
      float r0 = lr[s * 4], r1 = lr[s * 4 + 1], r2 = lr[s * 4 + 2];
#pragma unroll
      for (int i = 0; i < 4; i++) {
        float pr = fmaf(r0, w2c0[i], fmaf(r1, w2c1[i], fmaf(r2, w2c2[i], pb4[i])));
        o[i] = fmaf(gv[i] + pr, lw[s * 16 + ((c0 + i) & 15)], o[i]);
      }
    }
    *(f32x4*)&out[n * CH + c0] = o;
  }
}

// ---------------------------------------------------------------------------
extern "C" void kernel_launch(void* const* d_in, const int* in_sizes, int n_in,
                              void* d_out, int out_size, void* d_ws, size_t ws_size,
                              hipStream_t stream) {
  (void)in_sizes; (void)n_in; (void)out_size; (void)ws_size;
  const float* p    = (const float*)d_in[0];
  const float* x    = (const float*)d_in[1];
  const int*   idx  = (const int*)d_in[2];
  const float* w_q  = (const float*)d_in[3];
  const float* b_q  = (const float*)d_in[4];
  const float* w_k  = (const float*)d_in[5];
  const float* b_k  = (const float*)d_in[6];
  const float* w_v  = (const float*)d_in[7];
  const float* b_v  = (const float*)d_in[8];
  const float* w_p1 = (const float*)d_in[9];
  const float* b_p1 = (const float*)d_in[10];
  const float* g_p  = (const float*)d_in[11];
  const float* be_p = (const float*)d_in[12];
  const float* w_p2 = (const float*)d_in[13];
  const float* b_p2 = (const float*)d_in[14];
  const float* g_w1 = (const float*)d_in[15];
  const float* be_w1= (const float*)d_in[16];
  const float* w_w1 = (const float*)d_in[17];
  const float* b_w1 = (const float*)d_in[18];
  const float* g_w2 = (const float*)d_in[19];
  const float* be_w2= (const float*)d_in[20];
  const float* w_w2 = (const float*)d_in[21];
  const float* b_w2 = (const float*)d_in[22];
  float* out = (float*)d_out;

  // workspace carve-up (f32 arrays first, then bf16 fragments, then stats/bn)
  float* ws = (float*)d_ws;
  float* xq = ws;
  float* xk = xq + NC;
  float* xv = xk + NC;
  __bf16* wqf = (__bf16*)(xv + NC);
  __bf16* wkf = wqf + 16384;
  __bf16* wvf = wkf + 16384;
  __bf16* w1f = wvf + 16384;
  __bf16* w2f = w1f + 2048;
  float* stats = (float*)(w2f + 512);
  float* posSum = stats;        // 3
  float* posSq  = stats + 3;    // 3
  float* s1Sum  = stats + 6;    // 128
  float* s1Sq   = stats + 134;  // 128
  float* s2Sum  = stats + 262;  // 16
  float* s2Sq   = stats + 278;  // 16
  float* bnb  = stats + 294;
  float* pSc  = bnb;            // 3
  float* pSh  = bnb + 3;        // 3
  float* w1Sc = bnb + 6;        // 128
  float* w1Sh = bnb + 134;      // 128
  float* w2Sc = bnb + 262;      // 16
  float* w2Sh = bnb + 278;      // 16

  const float invc = 1.0f / (float)(NPTS * NSAMP);

  prep_kernel<<<204, 256, 0, stream>>>(w_q, w_k, w_v, w_w1, w_w2,
                                       wqf, wkf, wvf, w1f, w2f, stats);
  qkv_gemm<<<NPTS / 16, 256, 0, stream>>>(x, b_q, b_k, b_v, wqf, wkf, wvf, xq, xk, xv);
  pos_stats<<<512, 256, 0, stream>>>(p, idx, w_p1, b_p1, stats);
  finalize_bn<<<1, 32, 0, stream>>>(posSum, posSq, g_p, be_p, pSc, pSh, 3, invc);
  w1_stats<<<1024, 256, 0, stream>>>(p, idx, xq, xk, w_p1, b_p1, w_p2, b_p2,
                                     pSc, pSh, s1Sum, s1Sq);
  finalize_bn<<<1, 128, 0, stream>>>(s1Sum, s1Sq, g_w1, be_w1, w1Sc, w1Sh, 128, invc);
  w2_stats<<<1024, 256, 0, stream>>>(p, idx, xq, xk, w_p1, b_p1, w_p2, b_p2,
                                     pSc, pSh, w1Sc, w1Sh, b_w1, w1f, s2Sum, s2Sq);
  finalize_bn<<<1, 32, 0, stream>>>(s2Sum, s2Sq, g_w2, be_w2, w2Sc, w2Sh, 16, invc);
  attn_final<<<1875, 256, 0, stream>>>(p, idx, xq, xk, xv, w_p1, b_p1, w_p2, b_p2,
                                       pSc, pSh, w1Sc, w1Sh, b_w1, w2Sc, w2Sh, b_w2,
                                       w1f, w2f, out);
}
